// TransformerDWConv2d_87608742903946
// MI455X (gfx1250) — matrix operand order repaired and checked by reading, not compiled, not run on hardware
//
#include <hip/hip_runtime.h>
#include <hip/hip_bf16.h>
#include <math.h>

// ---- problem constants (match reference) ----
#define BB   2
#define CC   32
#define HH   96
#define WW   96
#define LL   (HH * WW)        // 9216
#define EE   12
#define NHEAD 4
#define DH   3
#define KK   3
#define FF   24
#define NSEQ (BB * LL * CC)   // 589824

// ---- tiling ----
#define SEQ_PER_WG 16
#define ROWS (SEQ_PER_WG * 9) // 144 token-rows per workgroup
#define MTILES 9              // 144 / 16
#define NWAVES 8              // 256 threads / wave32
#define SAST 40               // padded row stride (halves) of f16 A staging

// Weight B-tiles staged in LDS in WMMA lane layout:
//  tiles 0..2 : qkv (in_proj_w,  inDim=12, outW=36)
//  tile  3    : out_proj (12 -> 12)
//  tiles 4..5 : ff1 (lin1_w, 12 -> 24)
//  tile  6    : ff2 (lin2_w, 24 -> 12)
#define NBTILES 7
#define TQKV 0
#define TOUT 3
#define TFF1 4
#define TFF2 6

typedef __attribute__((ext_vector_type(16))) _Float16 v16h;
typedef __attribute__((ext_vector_type(8)))  _Float16 v8h;
typedef __attribute__((ext_vector_type(8)))  float    v8f;

// A-matrix lane layout (16-bit A 16x32, ISA 7.12.2):
//  lane<16 : halves 0..7 -> K=0..7,  halves 8..15 -> K=16..23, M=lane
//  lane>=16: halves 0..7 -> K=8..15, halves 8..15 -> K=24..31, M=lane-16
__device__ __forceinline__ v16h load_A(const _Float16* __restrict__ sA, int tileM, int lane) {
  const int row = tileM * 16 + (lane & 15);
  const _Float16* p = sA + row * SAST;
  const int k0 = (lane < 16) ? 0 : 8;
  const int k1 = (lane < 16) ? 16 : 24;
  v8h lo = *(const v8h*)(p + k0);   // 16B aligned chunks
  v8h hi = *(const v8h*)(p + k1);
  return __builtin_shufflevector(lo, hi, 0, 1, 2, 3, 4, 5, 6, 7,
                                 8, 9, 10, 11, 12, 13, 14, 15);
}

// D (f32 16x16, 8 VGPRs): lane<16 -> N=lane, M=r ; lane>=16 -> N=lane-16, M=8+r
__device__ __forceinline__ void store_D(float* __restrict__ dst, int ldd,
                                        const float* __restrict__ sBias, int tileB,
                                        int outW, int tileM, int tileN, int lane, v8f d) {
  const int n = tileN * 16 + (lane & 15);
  if (n >= outW) return;
  const float bv = sBias[tileB * 16 + (lane & 15)];
  const int mBase = (lane < 16) ? 0 : 8;
#pragma unroll
  for (int r = 0; r < 8; ++r) {
    const int row = tileM * 16 + mBase + r;
    dst[row * ldd + n] = d[r] + bv;
  }
}

__device__ __forceinline__ void wmma_gemm(const _Float16* __restrict__ sA,
                                          const _Float16* __restrict__ sB,
                                          const float* __restrict__ sBias,
                                          int tileBase, int nTilesN,
                                          float* __restrict__ dst, int ldd, int outW,
                                          int wave, int lane) {
  const int total = MTILES * nTilesN;
  for (int t = wave; t < total; t += NWAVES) {
    const int tm = t / nTilesN;
    const int tn = t % nTilesN;
    v16h a = load_A(sA, tm, lane);
    v16h b = *(const v16h*)&sB[(tileBase + tn) * 512 + lane * 16]; // 32B contiguous
    v8f  c = {};
    c = __builtin_amdgcn_wmma_f32_16x16x32_f16(false, a, false, b, (short)0, c, false, false);
    store_D(dst, ldd, sBias, tileBase + tn, outW, tm, tn, lane, c);
  }
}

__global__ __launch_bounds__(256)
void TransformerDWConv2d_gfx1250_kernel(
    const float* __restrict__ x,          // [B,C,H,W]
    const float* __restrict__ ch_emb,     // [C,E]
    const float* __restrict__ pos_emb,    // [9,E]
    const float* __restrict__ in_proj_w,  // [3E,E]
    const float* __restrict__ in_proj_b,  // [3E]
    const float* __restrict__ out_proj_w, // [E,E]
    const float* __restrict__ out_proj_b, // [E]
    const float* __restrict__ ln1_g, const float* __restrict__ ln1_b,
    const float* __restrict__ lin1_w,     // [FF,E]
    const float* __restrict__ lin1_b,     // [FF]
    const float* __restrict__ lin2_w,     // [E,FF]
    const float* __restrict__ lin2_b,     // [E]
    const float* __restrict__ ln2_g, const float* __restrict__ ln2_b,
    const float* __restrict__ head_w,     // [1,E]
    const float* __restrict__ head_b,     // [1]
    float* __restrict__ out)              // [B,C,H,W] (pre-zeroed)
{
  __shared__ __align__(32) _Float16 sA[ROWS * SAST];     // f16 A staging, reused per GEMM
  __shared__ __align__(32) _Float16 sB[NBTILES * 512];   // weight B tiles, WMMA lane layout
  __shared__ float sBias[NBTILES * 16];
  __shared__ float sSeq[ROWS * EE];   // embedding rows (residual 1)
  __shared__ float sQKV[ROWS * 36];   // qkv
  __shared__ float sAcc[ROWS * FF];   // gemm outputs (reused)
  __shared__ float sH1[ROWS * EE];    // post-LN1 (residual 2)

  const int tid  = threadIdx.x;
  const int lane = tid & 31;
  const int wave = tid >> 5;
  const int wg   = blockIdx.x;

  // ---------------- Phase 0a: stage weight B-tiles + biases into LDS ----------------
  for (int idx = tid; idx < NBTILES * 512; idx += 256) {
    const int t  = idx >> 9;         // tile
    const int ln = (idx >> 4) & 31;  // lane slot
    const int e  = idx & 15;         // half slot
    const float* Wt; int inDim, outW, tn;
    if (t < 3)       { Wt = in_proj_w;  inDim = EE; outW = 36; tn = t;     }
    else if (t == 3) { Wt = out_proj_w; inDim = EE; outW = EE; tn = 0;     }
    else if (t < 6)  { Wt = lin1_w;     inDim = EE; outW = FF; tn = t - 4; }
    else             { Wt = lin2_w;     inDim = FF; outW = EE; tn = 0;     }
    const int k = ((ln < 16) ? 0 : 8) + ((e < 8) ? e : e + 8);      // B: lane<16 -> K=0..7,16..23; lane>=16 -> K=8..15,24..31
    const int n = tn * 16 + (ln & 15);
    const float v = (k < inDim && n < outW) ? Wt[n * inDim + k] : 0.0f;
    sB[idx] = (_Float16)v;
  }
  for (int idx = tid; idx < NBTILES * 16; idx += 256) {
    const int t = idx >> 4, col = idx & 15;
    const float* bs; int outW, tn;
    if (t < 3)       { bs = in_proj_b;  outW = 36; tn = t;     }
    else if (t == 3) { bs = out_proj_b; outW = EE; tn = 0;     }
    else if (t < 6)  { bs = lin1_b;     outW = FF; tn = t - 4; }
    else             { bs = lin2_b;     outW = EE; tn = 0;     }
    const int n = tn * 16 + col;
    sBias[idx] = (n < outW) ? bs[n] : 0.0f;
  }

  // ---------------- Phase 0b: build sequence rows ----------------
  // row r = s*9 + t ; global sequence n = wg*16 + s ; n = ((b*L + l)*C + c)
  for (int idx = tid; idx < ROWS * 32; idx += 256) {
    const int r   = idx >> 5;
    const int col = idx & 31;
    const int s = r / 9, t = r % 9;
    const int n = wg * SEQ_PER_WG + s;
    const int c   = n % CC;
    const int rem = n / CC;
    const int l   = rem % LL;
    const int b   = rem / LL;
    const int hh  = l / WW, ww = l % WW;
    float v = 0.0f;
    if (col < EE) {
      const int i = t / KK, j = t % KK;
      const int yy = hh + i - 1, xx = ww + j - 1;
      float xv = 0.0f;
      if (yy >= 0 && yy < HH && xx >= 0 && xx < WW)
        xv = x[((b * CC + c) * HH + yy) * WW + xx];
      v = ch_emb[c * EE + col] + pos_emb[t * EE + col] + xv;
      sSeq[r * EE + col] = v;
    }
    sA[r * SAST + col] = (_Float16)v; // cols 12..31 zero padding
  }
  __syncthreads();

  // ---------------- Phase 1: QKV = seq @ in_proj_w^T + b ----------------
  wmma_gemm(sA, sB, sBias, TQKV, 3, sQKV, 36, 36, wave, lane);
  __syncthreads();

  // ---------------- Phase 2: attention, one thread per (seq, query token) ----------------
  if (tid < ROWS) {
    const int s  = tid / 9;
    const int tq = tid % 9;
    const int base = s * 9;
    const float scale = 0.57735026919f; // 1/sqrt(3)
    const float* qp = &sQKV[(base + tq) * 36];
#pragma unroll
    for (int hd = 0; hd < NHEAD; ++hd) {
      const float q0 = qp[hd * DH + 0];
      const float q1 = qp[hd * DH + 1];
      const float q2 = qp[hd * DH + 2];
      float sc[9], mx = -1e30f;
#pragma unroll
      for (int tk = 0; tk < 9; ++tk) {
        const float* kp = &sQKV[(base + tk) * 36 + EE + hd * DH];
        const float d = (q0 * kp[0] + q1 * kp[1] + q2 * kp[2]) * scale;
        sc[tk] = d;
        mx = fmaxf(mx, d);
      }
      float sum = 0.0f;
#pragma unroll
      for (int tk = 0; tk < 9; ++tk) { sc[tk] = __expf(sc[tk] - mx); sum += sc[tk]; }
      const float inv = 1.0f / sum;
      float o0 = 0, o1 = 0, o2 = 0;
#pragma unroll
      for (int tk = 0; tk < 9; ++tk) {
        const float* vp = &sQKV[(base + tk) * 36 + 2 * EE + hd * DH];
        const float a = sc[tk] * inv;
        o0 += a * vp[0]; o1 += a * vp[1]; o2 += a * vp[2];
      }
      sA[(base + tq) * SAST + hd * DH + 0] = (_Float16)o0;
      sA[(base + tq) * SAST + hd * DH + 1] = (_Float16)o1;
      sA[(base + tq) * SAST + hd * DH + 2] = (_Float16)o2;
      // cols 12..31 still zero from Phase 0b
    }
  }
  __syncthreads();

  // ---------------- Phase 3: out_proj GEMM ----------------
  wmma_gemm(sA, sB, sBias, TOUT, 1, sAcc, FF, EE, wave, lane);
  __syncthreads();

  // ---------------- Phase 4: h1 = LN1(seq + o) ----------------
  if (tid < ROWS) {
    const int row = tid;
    float v[EE], m = 0.0f;
#pragma unroll
    for (int e = 0; e < EE; ++e) { v[e] = sSeq[row * EE + e] + sAcc[row * FF + e]; m += v[e]; }
    m *= (1.0f / EE);
    float var = 0.0f;
#pragma unroll
    for (int e = 0; e < EE; ++e) { float d = v[e] - m; var += d * d; }
    var *= (1.0f / EE);
    const float rs = rsqrtf(var + 1e-5f);
#pragma unroll
    for (int e = 0; e < EE; ++e) {
      const float h = (v[e] - m) * rs * ln1_g[e] + ln1_b[e];
      sH1[row * EE + e] = h;
      sA[row * SAST + e] = (_Float16)h;
    }
#pragma unroll
    for (int e = EE; e < 32; ++e) sA[row * SAST + e] = (_Float16)0.0f;
  }
  __syncthreads();

  // ---------------- Phase 5: ff1 = h1 @ lin1_w^T + b ----------------
  wmma_gemm(sA, sB, sBias, TFF1, 2, sAcc, FF, FF, wave, lane);
  __syncthreads();

  // ---------------- Phase 6: relu restage ----------------
  if (tid < ROWS) {
    const int row = tid;
#pragma unroll
    for (int c = 0; c < 32; ++c) {
      const float v = (c < FF) ? fmaxf(sAcc[row * FF + c], 0.0f) : 0.0f;
      sA[row * SAST + c] = (_Float16)v;
    }
  }
  __syncthreads();

  // ---------------- Phase 7: ff2 = relu(ff1) @ lin2_w^T + b ----------------
  wmma_gemm(sA, sB, sBias, TFF2, 1, sAcc, FF, EE, wave, lane);
  __syncthreads();

  // ---------------- Phase 8: h2 = LN2(h1 + ff2); head; fold ----------------
  if (tid < ROWS) {
    const int row = tid;
    float v[EE], m = 0.0f;
#pragma unroll
    for (int e = 0; e < EE; ++e) { v[e] = sH1[row * EE + e] + sAcc[row * FF + e]; m += v[e]; }
    m *= (1.0f / EE);
    float var = 0.0f;
#pragma unroll
    for (int e = 0; e < EE; ++e) { float d = v[e] - m; var += d * d; }
    var *= (1.0f / EE);
    const float rs = rsqrtf(var + 1e-5f);
    float y = head_b[0];
#pragma unroll
    for (int e = 0; e < EE; ++e) {
      const float h2 = (v[e] - m) * rs * ln2_g[e] + ln2_b[e];
      y += h2 * head_w[e];
    }
    // fold: scatter-add into output
    const int s = row / 9, t = row % 9;
    const int n = wg * SEQ_PER_WG + s;
    const int c   = n % CC;
    const int rem = n / CC;
    const int l   = rem % LL;
    const int b   = rem / LL;
    const int hh = l / WW, ww = l % WW;
    const int i = t / KK, j = t % KK;
    const int yy = hh + i - 1, xx = ww + j - 1;
    if (yy >= 0 && yy < HH && xx >= 0 && xx < WW)
      atomicAdd(&out[((b * CC + c) * HH + yy) * WW + xx], y);
  }
}

extern "C" void kernel_launch(void* const* d_in, const int* in_sizes, int n_in,
                              void* d_out, int out_size, void* d_ws, size_t ws_size,
                              hipStream_t stream) {
  (void)in_sizes; (void)n_in; (void)d_ws; (void)ws_size;
  const float* x          = (const float*)d_in[0];
  const float* ch_emb     = (const float*)d_in[1];
  const float* pos_emb    = (const float*)d_in[2];
  const float* in_proj_w  = (const float*)d_in[3];
  const float* in_proj_b  = (const float*)d_in[4];
  const float* out_proj_w = (const float*)d_in[5];
  const float* out_proj_b = (const float*)d_in[6];
  const float* ln1_g      = (const float*)d_in[7];
  const float* ln1_b      = (const float*)d_in[8];
  const float* lin1_w     = (const float*)d_in[9];
  const float* lin1_b     = (const float*)d_in[10];
  const float* lin2_w     = (const float*)d_in[11];
  const float* lin2_b     = (const float*)d_in[12];
  const float* ln2_g      = (const float*)d_in[13];
  const float* ln2_b      = (const float*)d_in[14];
  const float* head_w     = (const float*)d_in[15];
  const float* head_b     = (const float*)d_in[16];
  float* out = (float*)d_out;

  hipMemsetAsync(out, 0, (size_t)out_size * sizeof(float), stream);

  const int grid = NSEQ / SEQ_PER_WG; // 36864 workgroups, 16 sequences each
  TransformerDWConv2d_gfx1250_kernel<<<grid, 256, 0, stream>>>(
      x, ch_emb, pos_emb, in_proj_w, in_proj_b, out_proj_w, out_proj_b,
      ln1_g, ln1_b, lin1_w, lin1_b, lin2_w, lin2_b, ln2_g, ln2_b,
      head_w, head_b, out);
}
